// Model_21629455303101
// MI455X (gfx1250) — matrix operand order repaired and checked by reading, not compiled, not run on hardware
//
#include <hip/hip_runtime.h>
#include <math.h>

#define BB 8
#define NN 4096
#define LL 256
#define EE 64
#define DD 128

typedef __attribute__((ext_vector_type(16))) __bf16 v16bf;
typedef __attribute__((ext_vector_type(8)))  float  v8f;
typedef __attribute__((address_space(3))) __bf16 lds_bf16_t;

// ---------- quaternion W(o,c) element: W row-block table ----------
// W = [[Qr,-Qi,-Qj,-Qk],[Qi,Qr,-Qk,Qj],[Qj,Qk,Qr,-Qi],[Qk,-Qj,Qi,Qr]]
__device__ __forceinline__ float quatW(int o, int c,
                                       const float* Qr, const float* Qi,
                                       const float* Qj, const float* Qk) {
  int bo = o >> 5, ro = o & 31, bc = c >> 5, rc = c & 31;
  const float* M; float s = 1.0f;
  switch (bo * 4 + bc) {
    case 0:  M = Qr; break;
    case 1:  M = Qi; s = -1.f; break;
    case 2:  M = Qj; s = -1.f; break;
    case 3:  M = Qk; s = -1.f; break;
    case 4:  M = Qi; break;
    case 5:  M = Qr; break;
    case 6:  M = Qk; s = -1.f; break;
    case 7:  M = Qj; break;
    case 8:  M = Qj; break;
    case 9:  M = Qk; break;
    case 10: M = Qr; break;
    case 11: M = Qi; s = -1.f; break;
    case 12: M = Qk; break;
    case 13: M = Qj; s = -1.f; break;
    case 14: M = Qi; break;
    default: M = Qr; break;
  }
  return s * M[ro * 32 + rc];
}

// ---------- t_he[b,l,d] = sin(mark[b,l]*W_temp[d] + b_temp[d]) ----------
__global__ void k_the(const float* __restrict__ mark, const float* __restrict__ Wt,
                      const float* __restrict__ bt, float* __restrict__ the) {
  int idx = blockIdx.x * 256 + threadIdx.x;           // B*L*D = 262144
  int d = idx & (DD - 1);
  int bl = idx >> 7;
  the[idx] = sinf(mark[bl] * Wt[d] + bt[d]);
}

__global__ void k_zero(float* __restrict__ p, int n) {
  int i = blockIdx.x * 256 + threadIdx.x;
  if (i < n) p[i] = 0.0f;
}

// ---------- counts[b,e] = sum_n mask^2 where var_idx==e ----------
__global__ void k_counts(const float* __restrict__ xmask, const int* __restrict__ vidx,
                         float* __restrict__ counts) {
  int i = blockIdx.x * 256 + threadIdx.x;             // B*N
  int b = i >> 12;
  float m = xmask[i];
  if (m != 0.0f) atomicAdd(&counts[b * EE + vidx[i]], m * m);
}

// ---------- q/k/v = relu(var_he_w) @ W + b  (batch-independent, 64x128) ----------
__global__ void k_qkv(const float* __restrict__ vhe,
                      const float* __restrict__ Wq, const float* __restrict__ bq,
                      const float* __restrict__ Wk, const float* __restrict__ bk,
                      const float* __restrict__ Wv, const float* __restrict__ bv,
                      float* __restrict__ q, float* __restrict__ k, float* __restrict__ v) {
  const float* W; const float* bias; float* dst;
  if (blockIdx.x == 0)      { W = Wq; bias = bq; dst = q; }
  else if (blockIdx.x == 1) { W = Wk; bias = bk; dst = k; }
  else                      { W = Wv; bias = bv; dst = v; }
  for (int o = threadIdx.x; o < EE * DD; o += blockDim.x) {
    int e = o >> 7, d = o & (DD - 1);
    float acc = bias[d];
    for (int c = 0; c < DD; ++c) {
      float h = vhe[e * DD + c];
      h = h > 0.0f ? h : 0.0f;
      acc += h * W[c * DD + d];
    }
    dst[o] = acc;
  }
}

// ---------- scores[e,f] = q[e].k[f]/sqrt(D)  (64x64) ----------
__global__ void k_scores(const float* __restrict__ q, const float* __restrict__ k,
                         float* __restrict__ sc) {
  int o = blockIdx.x * 256 + threadIdx.x;             // 4096
  int e = o >> 6, f = o & 63;
  float acc = 0.0f;
  for (int c = 0; c < DD; ++c) acc += q[e * DD + c] * k[f * DD + c];
  sc[o] = acc * 0.08838834764831845f;                 // 1/sqrt(128)
}

// ---------- per-batch softmax (diagonal blend) + attn @ v ----------
__global__ void k_attn(const float* __restrict__ sc, const float* __restrict__ counts,
                       const float* __restrict__ v, const float* __restrict__ thr_p,
                       const float* __restrict__ mc_p, float* __restrict__ vatt) {
  int b = blockIdx.x;          // 8
  int e = threadIdx.x;         // 64
  float cnt = counts[b * EE + e];
  float thr = thr_p[0], mcf = mc_p[0];
  float se = sc[e * EE + e];
  if (cnt != 0.0f && se > thr)
    se = (1.0f - mcf) * se + mcf * (cnt * (1.0f / 64.0f));   // aux = cnt/sqrt(N)
  float s[EE];
  #pragma unroll
  for (int f = 0; f < EE; ++f) {
    float x = sc[e * EE + f];
    s[f] = (f == e) ? se : x;
  }
  float mx = -3.4e38f;
  #pragma unroll
  for (int f = 0; f < EE; ++f) mx = fmaxf(mx, s[f]);
  float sum = 0.0f;
  #pragma unroll
  for (int f = 0; f < EE; ++f) { s[f] = __expf(s[f] - mx); sum += s[f]; }
  float inv = 1.0f / sum;
  for (int d = 0; d < DD; ++d) {
    float acc = 0.0f;
    #pragma unroll
    for (int f = 0; f < EE; ++f) acc += s[f] * v[f * DD + d];
    vatt[((size_t)b * EE + e) * DD + d] = acc * inv;
  }
}

// ---------- MfT[o,a] = (P @ W^T)[a,o]  stored transposed, bf16 ----------
__global__ void k_mft(const float* __restrict__ Pr_w, const float* __restrict__ Pi_w,
                      const float* __restrict__ Pj_w, const float* __restrict__ Pk_w,
                      const float* __restrict__ Qr, const float* __restrict__ Qi,
                      const float* __restrict__ Qj, const float* __restrict__ Qk,
                      __bf16* __restrict__ MfT) {
  int idx = blockIdx.x * 256 + threadIdx.x;           // 128*384 = 49152
  int o = idx / 384;
  int a = idx - o * 384;
  float acc = 0.0f;
  for (int c = 0; c < 32; ++c)
    acc += Pr_w[a * 32 + c] * quatW(o, c, Qr, Qi, Qj, Qk);
  int g = a >> 7;                                     // which 128-segment
  int ar = a & 127;
  const float* Pg = (g == 0) ? Pi_w : (g == 1) ? Pj_w : Pk_w;
  int cbase = 32 * (g + 1);
  for (int c = 0; c < 32; ++c)
    acc += Pg[ar * 32 + c] * quatW(o, cbase + c, Qr, Qi, Qj, Qk);
  MfT[o * 384 + a] = (__bf16)acc;
}

// ---------- bias_comb[o] = qbias @ W^T + fusion_bias ----------
__global__ void k_bias(const float* __restrict__ Pr_b, const float* __restrict__ Pi_b,
                       const float* __restrict__ Pj_b, const float* __restrict__ Pk_b,
                       const float* __restrict__ Qr, const float* __restrict__ Qi,
                       const float* __restrict__ Qj, const float* __restrict__ Qk,
                       const float* __restrict__ fbias, float* __restrict__ bias_out) {
  int o = threadIdx.x;                                // 128
  float acc = fbias[o];
  for (int c = 0; c < DD; ++c) {
    float qb = (c < 32) ? Pr_b[c] : (c < 64) ? Pi_b[c - 32]
             : (c < 96) ? Pj_b[c - 64] : Pk_b[c - 96];
    acc += qb * quatW(o, c, Qr, Qi, Qj, Qk);
  }
  bias_out[o] = acc;
}

// ---------- fused main GEMM: out[b,n,:] = [obs|tg|vg] @ Mf + bias ----------
// block = 128 threads (4 waves), each wave owns one 16-node row tile (64 nodes/block).
// Mf^T staged in LDS in two 192-K halves (50 KB) via GLOBAL_LOAD_ASYNC_TO_LDS_B128
// (ASYNCcnt, no VGPR round-trip), A fragments built on the fly (obs recompute +
// t_he/v_he_att gathers), v_wmma_f32_16x16x32_bf16 accumulation.
__global__ void __launch_bounds__(128) k_fused(
    const float* __restrict__ xL, const float* __restrict__ xmask,
    const float* __restrict__ ymask, const int* __restrict__ tidx,
    const int* __restrict__ vidx, const float* __restrict__ Wobs,
    const float* __restrict__ bobs, const float* __restrict__ the,
    const float* __restrict__ vatt, const __bf16* __restrict__ MfT,
    const float* __restrict__ biasc, float* __restrict__ out) {
  extern __shared__ __bf16 BT[];                      // [128][200] bf16 = 50 KB
  const int b   = blockIdx.x >> 6;                    // 8 batches
  const int n0  = (blockIdx.x & 63) * 64;             // 64 nodes per block
  const int lane = threadIdx.x & 31;
  const int wave = threadIdx.x >> 5;
  const int row  = lane & 15;                         // M index / N index
  const int half = lane >> 4;

  // LDS byte offset of this thread's staging row (addrspace(3) ptrtoint)
  const unsigned lofs =
      (unsigned)(size_t)(lds_bf16_t*)(BT + threadIdx.x * 200);

  const int node = n0 + wave * 16 + row;
  const size_t g = (size_t)b * NN + node;
  const float m  = xmask[g];
  const float x0 = xL[g];
  const float x1 = 1.0f - m + ymask[g];
  const float* trow = the  + ((size_t)b * LL + tidx[g]) * DD;
  const float* vrow = vatt + ((size_t)b * EE + vidx[g]) * DD;

  v8f acc[8];
  #pragma unroll
  for (int ct = 0; ct < 8; ++ct)
    #pragma unroll
    for (int r = 0; r < 8; ++r) acc[ct][r] = 0.0f;

  #pragma unroll
  for (int ph = 0; ph < 2; ++ph) {
    __syncthreads();   // previous-phase readers done before restaging
    {   // async-stage one K-half of MfT: thread t copies column-row t (384 B)
        // INST_OFFSET is added to BOTH lds and global addresses (ISA 10.x),
        // so a stepping offset walks the contiguous row on both sides.
      const unsigned gofs = (unsigned)(threadIdx.x * 768 + ph * 384); // bytes
      #pragma unroll
      for (int i = 0; i < 24; ++i) {
        asm volatile("global_load_async_to_lds_b128 %0, %1, %2 offset:%3"
                     :
                     : "v"(lofs), "v"(gofs), "s"(MfT), "i"(i * 16)
                     : "memory");
      }
      asm volatile("s_wait_asynccnt 0" ::: "memory");
    }
    __syncthreads();
    #pragma unroll
    for (int kk = 0; kk < 6; ++kk) {
      const int k0 = (ph * 6 + kk) * 32;
      // ---- A fragment (16x32 bf16), ISA layout: e<8 -> K=half*8+e ; e>=8 -> K=16+half*8+(e-8)
      v16bf af;
      #pragma unroll
      for (int e = 0; e < 16; ++e) {
        const int kl = (e < 8) ? (half * 8 + e) : (8 + half * 8 + e);
        const int f = k0 + kl;                         // segment resolved at compile time
        float val;
        if (f < DD) {
          val = fmaxf(x0 * Wobs[f] + x1 * Wobs[DD + f] + bobs[f], 0.0f);
        } else if (f < 2 * DD) {
          val = trow[f - DD];
        } else {
          val = vrow[f - 2 * DD];
        }
        af[e] = (__bf16)(val * m);
      }
      // ---- 8 column tiles, B frag: lane col = row, e<8 -> K=half*8+e ; e>=8 -> K=16+half*8+(e-8)
      #pragma unroll
      for (int ct = 0; ct < 8; ++ct) {
        const __bf16* Brow = BT + (ct * 16 + row) * 200 + kk * 32 + half * 8;
        v16bf bfr;
        #pragma unroll
        for (int e2 = 0; e2 < 16; ++e2) bfr[e2] = Brow[(e2 < 8) ? e2 : (e2 + 8)];
        acc[ct] = __builtin_amdgcn_wmma_f32_16x16x32_bf16(
            false, af, false, bfr, (short)0, acc[ct], false, false);
      }
    }
  }

  // ---- epilogue: C/D layout N=row, M=half*8+r
  const size_t ob = ((size_t)b * NN + n0 + wave * 16) * DD;
  #pragma unroll
  for (int ct = 0; ct < 8; ++ct) {
    const int colg = ct * 16 + row;
    const float bv = biasc[colg];
    #pragma unroll
    for (int r = 0; r < 8; ++r)
      out[ob + (size_t)(half * 8 + r) * DD + colg] = acc[ct][r] + bv;
  }
}

extern "C" void kernel_launch(void* const* d_in, const int* in_sizes, int n_in,
                              void* d_out, int out_size, void* d_ws, size_t ws_size,
                              hipStream_t stream) {
  const float* x_L    = (const float*)d_in[0];
  const float* xmask  = (const float*)d_in[1];
  const float* ymask  = (const float*)d_in[2];
  const float* mark   = (const float*)d_in[3];
  const int*   vidx   = (const int*)d_in[4];
  const int*   tidx   = (const int*)d_in[5];
  // d_in[6] = N_OBS_MAX (unused, equals N)
  const float* W_obs  = (const float*)d_in[7];
  const float* b_obs  = (const float*)d_in[8];
  const float* W_temp = (const float*)d_in[9];
  const float* b_temp = (const float*)d_in[10];
  const float* var_he = (const float*)d_in[11];
  const float* Wq = (const float*)d_in[12]; const float* bq = (const float*)d_in[13];
  const float* Wk = (const float*)d_in[14]; const float* bk = (const float*)d_in[15];
  const float* Wv = (const float*)d_in[16]; const float* bv = (const float*)d_in[17];
  const float* thr = (const float*)d_in[18];
  const float* mc  = (const float*)d_in[19];
  const float* Pr_w = (const float*)d_in[20]; const float* Pr_b = (const float*)d_in[21];
  const float* Pi_w = (const float*)d_in[22]; const float* Pi_b = (const float*)d_in[23];
  const float* Pj_w = (const float*)d_in[24]; const float* Pj_b = (const float*)d_in[25];
  const float* Pk_w = (const float*)d_in[26]; const float* Pk_b = (const float*)d_in[27];
  const float* Qr = (const float*)d_in[28];
  const float* Qi = (const float*)d_in[29];
  const float* Qj = (const float*)d_in[30];
  const float* Qk = (const float*)d_in[31];
  const float* fbias = (const float*)d_in[32];

  char* ws = (char*)d_ws;
  float*  ws_the    = (float*)(ws);                    // 8*256*128*4 = 1048576
  float*  ws_vatt   = (float*)(ws + 1048576);          // 8*64*128*4  = 262144
  float*  ws_counts = (float*)(ws + 1310720);          // 8*64*4      = 2048
  float*  ws_q      = (float*)(ws + 1312768);          // 64*128*4    = 32768
  float*  ws_k      = (float*)(ws + 1345536);
  float*  ws_v      = (float*)(ws + 1378304);
  float*  ws_sc     = (float*)(ws + 1411072);          // 64*64*4     = 16384
  __bf16* ws_MfT    = (__bf16*)(ws + 1427456);         // 128*384*2   = 98304
  float*  ws_bias   = (float*)(ws + 1525760);          // 128*4

  k_the   <<<dim3(1024), dim3(256), 0, stream>>>(mark, W_temp, b_temp, ws_the);
  k_zero  <<<dim3(2),    dim3(256), 0, stream>>>(ws_counts, BB * EE);
  k_counts<<<dim3(128),  dim3(256), 0, stream>>>(xmask, vidx, ws_counts);
  k_qkv   <<<dim3(3),    dim3(256), 0, stream>>>(var_he, Wq, bq, Wk, bk, Wv, bv,
                                                 ws_q, ws_k, ws_v);
  k_scores<<<dim3(16),   dim3(256), 0, stream>>>(ws_q, ws_k, ws_sc);
  k_attn  <<<dim3(8),    dim3(64),  0, stream>>>(ws_sc, ws_counts, ws_v, thr, mc, ws_vatt);
  k_mft   <<<dim3(192),  dim3(256), 0, stream>>>(Pr_w, Pi_w, Pj_w, Pk_w,
                                                 Qr, Qi, Qj, Qk, ws_MfT);
  k_bias  <<<dim3(1),    dim3(128), 0, stream>>>(Pr_b, Pi_b, Pj_b, Pk_b,
                                                 Qr, Qi, Qj, Qk, fbias, ws_bias);
  k_fused <<<dim3(512),  dim3(128), 128 * 200 * 2, stream>>>(
      x_L, xmask, ymask, tidx, vidx, W_obs, b_obs,
      ws_the, ws_vatt, ws_MfT, ws_bias, (float*)d_out);
}
